// MultiHeadAttentioin_13099650253106
// MI455X (gfx1250) — matrix operand order repaired and checked by reading, not compiled, not run on hardware
//
#include <hip/hip_runtime.h>
#include <hip/hip_bf16.h>

// ---------------------------------------------------------------------------
// MI455X (gfx1250) multi-head attention, bf16 WMMA pipeline.
//   B=4, S=1024, D_MODEL=1024, H=16, D_K=64
// Workspace (bf16 = ushort), 32 MB total:
//   Qws [B][H][S][Dk] | Kws [B][H][S][Dk] | Vws [B][H][Dk][S] | Aws [B][S][1024]
// ---------------------------------------------------------------------------

#define D_MODEL 1024
#define HEADS   16
#define DK      64
#define BATCH   4
#define SEQ     1024

typedef __attribute__((ext_vector_type(16))) __bf16       bf16x16;
typedef __attribute__((ext_vector_type(8)))  float        f32x8;
typedef __attribute__((ext_vector_type(4)))  float        f32x4;
typedef __attribute__((ext_vector_type(4)))  unsigned int u32x4;

union FragB16 {          // one 16x32 (A) or 32x16 (B) bf16 WMMA operand
    bf16x16 v;
    u32x4   q[2];
};

__device__ __forceinline__ unsigned short f2bf(float f) {
    union { float f; unsigned u; } c; c.f = f;
    unsigned u = c.u;
    u += 0x7FFFu + ((u >> 16) & 1u);     // round-to-nearest-even
    return (unsigned short)(u >> 16);
}

__device__ __forceinline__ f32x8 zero8() {
    f32x8 z;
    #pragma unroll
    for (int i = 0; i < 8; i++) z[i] = 0.0f;
    return z;
}

// v_permlane16_b32 butterfly max over a 16-lane row (pure VALU, no LDS traffic)
__device__ __forceinline__ float permlane_xor(float x, unsigned s1, unsigned s2) {
    unsigned u = __float_as_uint(x);
    unsigned r = __builtin_amdgcn_permlane16(u, u, s1, s2, false, false);
    return __uint_as_float(r);
}
__device__ __forceinline__ float rowmax16(float x) {
    x = fmaxf(x, permlane_xor(x, 0x67452301u, 0xEFCDAB89u));  // xor 1
    x = fmaxf(x, permlane_xor(x, 0x54761032u, 0xDCFE98BAu));  // xor 2
    x = fmaxf(x, permlane_xor(x, 0x32107654u, 0xBA98FEDCu));  // xor 4
    x = fmaxf(x, permlane_xor(x, 0xFEDCBA98u, 0x76543210u));  // xor 8
    return x;
}

// Async global->LDS copy engine (ASYNCcnt path, gfx1250).
// lds_off = addr[31:0] of a generic shared pointer (LDS aperture layout).
__device__ __forceinline__ void async_load_b128(unsigned lds_off, const void* gaddr) {
    asm volatile("global_load_async_to_lds_b128 %0, %1, off"
                 :: "v"(lds_off), "v"(gaddr) : "memory");
}
__device__ __forceinline__ void wait_asynccnt0() {
    asm volatile("s_wait_asynccnt 0" ::: "memory");
}

// ---------------------------------------------------------------------------
// Tiled GEMM: out(row,col) = sum_k A(row,k) * W(k,col) + bias(col)
// Double-buffered LDS (ping-pong), one barrier per 32-K step.
//   A fp32  (A_BF16=false): register-staged fetch + f32->bf16 convert.
//   A bf16  (A_BF16=true):  GLOBAL_LOAD_ASYNC_TO_LDS_B128 (no VGPR staging).
// Tile 128x128, 256 threads (8 waves), wave = 32x64.
// OUT_MODE 0: bf16 [b][h][s][d] (Q/K) | 1: bf16 [b][h][d][s] (V) | 2: fp32 flat
// ---------------------------------------------------------------------------
template<int OUT_MODE, bool A_BF16>
__global__ __launch_bounds__(256)
void gemm_wmma(const void* __restrict__ Aptr,
               const float* __restrict__ W,
               const float* __restrict__ bias,
               void* __restrict__ Out)
{
    __shared__ alignas(16) unsigned short As[2][128][32];   // [buf][m][k]
    __shared__ alignas(16) unsigned short Bs[2][128][32];   // [buf][n][k]

    const int tid   = threadIdx.x;
    const int lane  = tid & 31;
    const int wid   = tid >> 5;
    const int hi    = lane >> 4;
    const int ln16  = lane & 15;

    const int blockN = blockIdx.x * 128;
    const int blockM = blockIdx.y * 128;

    const int mBase = (wid >> 1) * 32;
    const int nBase = (wid & 1) * 64;

    f32x8 acc[2][4];
    #pragma unroll
    for (int i = 0; i < 2; i++)
        #pragma unroll
        for (int j = 0; j < 4; j++) acc[i][j] = zero8();

    const int aRow = tid >> 1;           // 0..127
    const int aK   = (tid & 1) * 16;     // 0 / 16
    const int wK   = tid & 31;           // 0..31
    const int wN   = (tid >> 5) * 16;    // 0..112

    const float*          Af = (const float*)Aptr;
    const unsigned short* Ab = (const unsigned short*)Aptr;

    unsigned short aReg[16];             // staged A slice (bf16, fp32 path only)
    unsigned short wReg[16];             // staged W slice (bf16)

    auto fetchA = [&](int k0) {
        const f32x4* src = (const f32x4*)(Af + (size_t)(blockM + aRow) * D_MODEL + k0 + aK);
        #pragma unroll
        for (int i = 0; i < 4; i++) {
            f32x4 f = src[i];
            aReg[i*4+0] = f2bf(f.x); aReg[i*4+1] = f2bf(f.y);
            aReg[i*4+2] = f2bf(f.z); aReg[i*4+3] = f2bf(f.w);
        }
    };
    auto asyncA = [&](int k0, int buf) {   // bf16 path: async copy, no staging
        const unsigned short* g = Ab + (size_t)(blockM + aRow) * D_MODEL + k0 + aK;
        const unsigned l0 = (unsigned)(size_t)&As[buf][aRow][aK];
        async_load_b128(l0,      g);
        async_load_b128(l0 + 16, g + 8);
    };
    auto fetchW = [&](int k0) {
        const f32x4* src = (const f32x4*)(W + (size_t)(k0 + wK) * D_MODEL + blockN + wN);
        #pragma unroll
        for (int i = 0; i < 4; i++) {
            f32x4 f = src[i];
            wReg[i*4+0] = f2bf(f.x); wReg[i*4+1] = f2bf(f.y);
            wReg[i*4+2] = f2bf(f.z); wReg[i*4+3] = f2bf(f.w);
        }
    };
    auto storeA = [&](int buf) {
        *(u32x4*)&As[buf][aRow][aK]     = *(u32x4*)&aReg[0];
        *(u32x4*)&As[buf][aRow][aK + 8] = *(u32x4*)&aReg[8];
    };
    auto storeW = [&](int buf) {
        #pragma unroll
        for (int i = 0; i < 16; i++)
            Bs[buf][wN + i][wK] = wReg[i];        // transpose scatter
    };

    // ---- pipeline prologue ----
    if constexpr (A_BF16) asyncA(0, 0); else fetchA(0);
    fetchW(0);
    if constexpr (!A_BF16) storeA(0);
    storeW(0);
    if constexpr (A_BF16) wait_asynccnt0();
    __syncthreads();

    constexpr int NIT = D_MODEL / 32;
    #pragma unroll 1
    for (int it = 0; it < NIT; it++) {
        const int buf = it & 1;
        if (it < NIT - 1) {                        // prefetch next tile
            if constexpr (A_BF16) asyncA((it + 1) * 32, buf ^ 1);
            else                  fetchA((it + 1) * 32);
            fetchW((it + 1) * 32);
        }

        // ---- WMMA: 2 A-fragments x 4 B-fragments on buf ----
        FragB16 afr[2];
        #pragma unroll
        for (int mt = 0; mt < 2; mt++) {
            const unsigned short* ap = &As[buf][mBase + mt*16 + ln16][0];
            afr[mt].q[0] = *(const u32x4*)(ap + hi*8);
            afr[mt].q[1] = *(const u32x4*)(ap + 16 + hi*8);
        }
        #pragma unroll
        for (int nt = 0; nt < 4; nt++) {
            FragB16 bfr;
            const unsigned short* bp = &Bs[buf][nBase + nt*16 + ln16][hi*8];
            bfr.q[0] = *(const u32x4*)(bp);
            bfr.q[1] = *(const u32x4*)(bp + 16);
            #pragma unroll
            for (int mt = 0; mt < 2; mt++) {
                acc[mt][nt] = __builtin_amdgcn_wmma_f32_16x16x32_bf16(
                    false, afr[mt].v, false, bfr.v,
                    (short)0, acc[mt][nt], false, false);
            }
        }

        if (it < NIT - 1) {
            if constexpr (!A_BF16) storeA(buf ^ 1);
            storeW(buf ^ 1);
            if constexpr (A_BF16) wait_asynccnt0();  // async tile landed in LDS
        }
        __syncthreads();                             // single barrier per K-step
    }

    // ---- epilogue: bias + layout-specific store ----
    #pragma unroll
    for (int nt = 0; nt < 4; nt++) {
        const int col = blockN + nBase + nt*16 + ln16;
        const float bv = bias[col];
        #pragma unroll
        for (int mt = 0; mt < 2; mt++) {
            #pragma unroll
            for (int r = 0; r < 8; r++) {
                const int row = blockM + mBase + mt*16 + hi*8 + r;
                const float val = acc[mt][nt][r] + bv;
                if (OUT_MODE == 2) {
                    ((float*)Out)[(size_t)row * D_MODEL + col] = val;
                } else {
                    const int b = row >> 10, s = row & 1023;
                    const int h = col >> 6,  d = col & 63;
                    size_t idx;
                    if (OUT_MODE == 0)
                        idx = (((size_t)(b*HEADS + h))*SEQ + s)*DK + d;
                    else
                        idx = (((size_t)(b*HEADS + h))*DK + d)*SEQ + s;
                    ((unsigned short*)Out)[idx] = f2bf(val);
                }
            }
        }
    }
}

// ---------------------------------------------------------------------------
// Flash attention: one block (4 waves) = one (b,h) x 64 query rows.
// 64-key chunks: 8 score WMMA + 8 PV WMMA + 2 row-sum WMMA (P @ ones).
// Row max: v_permlane16_b32 butterflies. Next-chunk K/V prefetched to cache.
// ---------------------------------------------------------------------------
__global__ __launch_bounds__(128)
void attention_wmma(const unsigned short* __restrict__ Q,
                    const unsigned short* __restrict__ K,
                    const unsigned short* __restrict__ Vt,
                    const int* __restrict__ mask,
                    unsigned short* __restrict__ Outh)
{
    __shared__ alignas(16) unsigned short Plds[4][16][64];  // per-wave P scratch

    const int tid  = threadIdx.x;
    const int lane = tid & 31;
    const int w    = tid >> 5;
    const int hi   = lane >> 4;
    const int ln16 = lane & 15;

    const int qt = blockIdx.x & 15;
    const int bh = blockIdx.x >> 4;
    const int b  = bh >> 4;
    const int h  = bh & 15;

    const int qBase = qt*64 + w*16;

    // Q A-fragments (16 rows x 64 d), loaded once from global.
    FragB16 qf[2];
    {
        const unsigned short* qp = Q + ((size_t)bh*SEQ + qBase + ln16)*DK;
        #pragma unroll
        for (int j = 0; j < 2; j++) {
            qf[j].q[0] = *(const u32x4*)(qp + j*32 + hi*8);
            qf[j].q[1] = *(const u32x4*)(qp + j*32 + 16 + hi*8);
        }
    }

    // ones B-matrix for row-sum WMMA (bf16 1.0 = 0x3F80)
    FragB16 ones;
    #pragma unroll
    for (int i = 0; i < 4; i++) { ones.q[0][i] = 0x3F803F80u; ones.q[1][i] = 0x3F803F80u; }

    f32x8 accO[4];          // O tile: 16 rows x 64 d
    f32x8 accL;             // running row-sum (replicated across columns)
    #pragma unroll
    for (int i = 0; i < 4; i++) accO[i] = zero8();
    accL = zero8();
    float mrow[8];
    #pragma unroll
    for (int r = 0; r < 8; r++) mrow[r] = -3.0e38f;

    #pragma unroll 1
    for (int kc = 0; kc < SEQ; kc += 64) {
        // ---- prefetch next chunk's K rows and V rows into cache ----
        if (kc + 64 < SEQ) {
            __builtin_prefetch(K  + ((size_t)bh*SEQ + kc + 64 + lane*2)*DK, 0, 2);
            __builtin_prefetch(Vt + ((size_t)bh*DK + lane*2)*SEQ + kc + 64, 0, 2);
        }

        // ---- scores: four 16x16 tiles over this 64-key chunk ----
        f32x8 sc[4];
        #pragma unroll
        for (int st = 0; st < 4; st++) {
            const int key = kc + st*16 + ln16;
            const unsigned short* kp = K + ((size_t)bh*SEQ + key)*DK + hi*8;
            f32x8 a = zero8();
            #pragma unroll
            for (int j = 0; j < 2; j++) {
                FragB16 bf;
                bf.q[0] = *(const u32x4*)(kp + j*32);
                bf.q[1] = *(const u32x4*)(kp + j*32 + 16);
                a = __builtin_amdgcn_wmma_f32_16x16x32_bf16(
                        false, qf[j].v, false, bf.v, (short)0, a, false, false);
            }
            const float mb = (mask[b*SEQ + key] == 0) ? -1.0e9f : 0.0f;
            #pragma unroll
            for (int r = 0; r < 8; r++)
                a[r] = fmaf(a[r], 0.125f, mb);   // scale 1/sqrt(64) + mask bias
            sc[st] = a;
        }

        // ---- online softmax: permlane16 max butterfly, exp ----
        float corr_r[8];
        #pragma unroll
        for (int r = 0; r < 8; r++) {
            float tmax = fmaxf(fmaxf(sc[0][r], sc[1][r]), fmaxf(sc[2][r], sc[3][r]));
            tmax = rowmax16(tmax);
            const float mnew = fmaxf(mrow[r], tmax);
            corr_r[r] = __expf(mrow[r] - mnew);
            mrow[r] = mnew;
            #pragma unroll
            for (int st = 0; st < 4; st++)
                sc[st][r] = __expf(sc[st][r] - mnew);
        }
        #pragma unroll
        for (int nt = 0; nt < 4; nt++)
            #pragma unroll
            for (int r = 0; r < 8; r++)
                accO[nt][r] *= corr_r[r];
        #pragma unroll
        for (int r = 0; r < 8; r++)
            accL[r] *= corr_r[r];

        // ---- P: C-layout -> LDS -> two A-layout fragments ----
        #pragma unroll
        for (int st = 0; st < 4; st++)
            #pragma unroll
            for (int r = 0; r < 8; r++)
                Plds[w][hi*8 + r][st*16 + ln16] = f2bf(sc[st][r]);
        asm volatile("s_wait_dscnt 0" ::: "memory");   // per-wave LDS RAW

        FragB16 pf[2];
        #pragma unroll
        for (int j = 0; j < 2; j++) {
            const unsigned short* pp = &Plds[w][ln16][j*32];
            pf[j].q[0] = *(const u32x4*)(pp + hi*8);
            pf[j].q[1] = *(const u32x4*)(pp + 16 + hi*8);
        }

        // ---- row sums via WMMA: accL += P @ ones ----
        #pragma unroll
        for (int j = 0; j < 2; j++)
            accL = __builtin_amdgcn_wmma_f32_16x16x32_bf16(
                       false, pf[j].v, false, ones.v, (short)0, accL, false, false);

        // ---- accO += P @ V ----
        #pragma unroll
        for (int nt = 0; nt < 4; nt++) {
            const int d = nt*16 + ln16;
            const unsigned short* vbase = Vt + ((size_t)bh*DK + d)*SEQ + kc;
            #pragma unroll
            for (int j = 0; j < 2; j++) {
                const unsigned short* vp = vbase + j*32 + hi*8;
                FragB16 vf;
                vf.q[0] = *(const u32x4*)(vp);
                vf.q[1] = *(const u32x4*)(vp + 16);
                accO[nt] = __builtin_amdgcn_wmma_f32_16x16x32_bf16(
                               false, pf[j].v, false, vf.v, (short)0, accO[nt], false, false);
            }
        }
    }

    // ---- normalize and store heads output: Aws[b][s][h*64+d] ----
    #pragma unroll
    for (int nt = 0; nt < 4; nt++) {
        const int d = nt*16 + ln16;
        #pragma unroll
        for (int r = 0; r < 8; r++) {
            const int srow = qBase + hi*8 + r;
            const float val = accO[nt][r] * (1.0f / accL[r]);
            Outh[((size_t)b*SEQ + srow)*D_MODEL + h*DK + d] = f2bf(val);
        }
    }
}

// ---------------------------------------------------------------------------
extern "C" void kernel_launch(void* const* d_in, const int* in_sizes, int n_in,
                              void* d_out, int out_size, void* d_ws, size_t ws_size,
                              hipStream_t stream)
{
    const float* q    = (const float*)d_in[0];
    const float* k    = (const float*)d_in[1];
    const float* v    = (const float*)d_in[2];
    const int*   mask = (const int*)  d_in[3];
    const float* W_Q  = (const float*)d_in[4];
    const float* b_Q  = (const float*)d_in[5];
    const float* W_K  = (const float*)d_in[6];
    const float* b_K  = (const float*)d_in[7];
    const float* W_V  = (const float*)d_in[8];
    const float* b_V  = (const float*)d_in[9];
    const float* W_O  = (const float*)d_in[10];
    const float* b_O  = (const float*)d_in[11];

    const size_t N_HEAD_ELEMS = (size_t)BATCH * HEADS * SEQ * DK;  // 4M
    unsigned short* Qws = (unsigned short*)d_ws;
    unsigned short* Kws = Qws + N_HEAD_ELEMS;
    unsigned short* Vws = Kws + N_HEAD_ELEMS;
    unsigned short* Aws = Vws + N_HEAD_ELEMS;   // 32 MB total

    dim3 grid(D_MODEL / 128, (BATCH * SEQ) / 128);   // 8 x 32
    dim3 block(256);

    gemm_wmma<0, false><<<grid, block, 0, stream>>>(q, W_Q, b_Q, Qws);
    gemm_wmma<0, false><<<grid, block, 0, stream>>>(k, W_K, b_K, Kws);
    gemm_wmma<1, false><<<grid, block, 0, stream>>>(v, W_V, b_V, Vws);

    attention_wmma<<<dim3(BATCH * HEADS * (SEQ / 64)), dim3(128), 0, stream>>>(
        Qws, Kws, Vws, mask, Aws);

    gemm_wmma<2, true><<<grid, block, 0, stream>>>(Aws, W_O, b_O, (float*)d_out);
}
